// Koopman_RNN_80092550136101
// MI455X (gfx1250) — matrix operand order repaired and checked by reading, not compiled, not run on hardware
//
#include <hip/hip_runtime.h>
#include <hip/hip_bf16.h>

// ---------------- problem constants ----------------
#define BB   128
#define TT   1024
#define DD   64
#define H1   256
#define H2   16
#define RR   8
#define PP   4
#define KAUX 128
#define DE   256
#define BT   (BB*TT)
#define DT_C 0.01f
#define RECON_HP_C 1.0f
#define NWG1 8          // persistent workgroups for LSTM1 (1024 cols / 128 per WG)
#define ASTR 328        // padded K-stride (elems) for A slab in LDS (656B -> 16-bank spread)
#define BSTR 328        // padded K-stride (elems) for transposed B slab in LDS

typedef __attribute__((ext_vector_type(16))) __bf16 bf16x16;
typedef __attribute__((ext_vector_type(8)))  __bf16 bf16x8;
typedef __attribute__((ext_vector_type(8)))  float  f32x8;
typedef __attribute__((ext_vector_type(4)))  unsigned int u32x4;
typedef __attribute__((ext_vector_type(4)))  int  i32x4;
typedef __attribute__((ext_vector_type(8)))  int  i32x8;

#if __has_builtin(__builtin_amdgcn_tensor_load_to_lds)
#define HAVE_TDM 1
#else
#define HAVE_TDM 0
#endif

static __device__ __forceinline__ unsigned short f2bf(float f) {
  unsigned u = __builtin_bit_cast(unsigned, f);
  unsigned r = (u + 0x7fffu + ((u >> 16) & 1u)) >> 16;   // RNE
  return (unsigned short)r;
}
static __device__ __forceinline__ bf16x16 cat8(bf16x8 a, bf16x8 b) {
  return __builtin_shufflevector(a, b, 0,1,2,3,4,5,6,7,8,9,10,11,12,13,14,15);
}
static __device__ __forceinline__ float sigf(float x) { return 1.f / (1.f + __expf(-x)); }
static __device__ __forceinline__ float leaky(float x) { return x > 0.f ? x : 0.3f * x; }

// ---------------- K0: fp32 -> bf16 prep (+W2 transpose) + sync init ----------------
__global__ void k_prep(const float* __restrict__ X, const float* __restrict__ W1,
                       const float* __restrict__ U1, const float* __restrict__ W2,
                       unsigned short* __restrict__ Xb, unsigned short* __restrict__ W1b,
                       unsigned short* __restrict__ U1b, unsigned short* __restrict__ W2bt,
                       unsigned* __restrict__ sync) {
  const long n0 = (long)BT * DD;        // inputs
  const long n1 = (long)DD * 4 * H1;    // W1
  const long n2 = (long)H1 * 4 * H1;    // U1
  const long n3 = (long)H1 * 4 * H2;    // W2
  const long total = n0 + n1 + n2 + n3;
  for (long i = (long)blockIdx.x * blockDim.x + threadIdx.x; i < total;
       i += (long)gridDim.x * blockDim.x) {
    if (i < n0)                Xb[i]            = f2bf(X[i]);
    else if (i < n0 + n1)      W1b[i - n0]      = f2bf(W1[i - n0]);
    else if (i < n0 + n1 + n2) U1b[i - n0 - n1] = f2bf(U1[i - n0 - n1]);
    else {
      long j = i - n0 - n1 - n2;                 // j = k*64 + c
      long k = j >> 6, c = j & 63;
      W2bt[c * H1 + k] = f2bf(W2[j]);            // K-contiguous per output column
    }
  }
  if (blockIdx.x == 0 && threadIdx.x < 2) sync[threadIdx.x] = 0u;
}

// ---------------- K1: persistent LSTM1 recurrence (WMMA bf16 + TDM restage) ----------------
// WG w owns hidden units [w*32, w*32+32): 128 local columns = {i,f,g,o} x 32.
// Per step: Z[128x128] = [x_t | h_{t-1}] (128x320 bf16) @ Bslab (320x128 bf16).
__global__ void __launch_bounds__(256, 1)
k_lstm1(const float* __restrict__ b1,
        const unsigned short* __restrict__ Xb,
        const unsigned short* __restrict__ W1b,
        const unsigned short* __restrict__ U1b,
        unsigned short* __restrict__ H1seq,   // (B,T,H1) bf16
        unsigned* __restrict__ sync) {
  extern __shared__ char smem[];
  unsigned short* Bs  = (unsigned short*)smem;          // [128 cols][BSTR] transposed (K-contig)
  unsigned short* As  = Bs + 128 * BSTR;                // [128 rows][ASTR]
  float*          Cs  = (float*)(As + 128 * ASTR);      // [128][32] cell state
  float*          b1s = Cs + 128 * 32;                  // [128]
  const unsigned AS_LDS_OFF = 128u * BSTR * 2u;         // byte offset of As in LDS

  const int w   = blockIdx.x;
  const int tid = threadIdx.x;

  // Stage weight slab once, K-contiguous per local column (resident for all 1024 steps).
  for (int i = tid; i < 128 * 320; i += 256) {
    int lc = i / 320, k = i % 320;
    int gcol = (lc >> 5) * H1 + w * 32 + (lc & 31);     // gate*256 + unit
    Bs[lc * BSTR + k] = (k < DD) ? W1b[k * (4 * H1) + gcol]
                                 : U1b[(k - DD) * (4 * H1) + gcol];
  }
  for (int i = tid; i < 128; i += 256)
    b1s[i] = b1[(i >> 5) * H1 + w * 32 + (i & 31)];
  for (int i = tid; i < 128 * 32; i += 256) Cs[i] = 0.f;
  __syncthreads();

  const int lane = tid & 31, wave = tid >> 5;
  const int half = lane >> 4, ln = lane & 15;
  const int rbase = wave * 16;                  // 8 waves x 16 rows = 128 batch rows
  const int arow = rbase + ln;

  for (int t = 0; t < TT; ++t) {
    // ---- Stage A = [x_t | h_{t-1}] (bf16, padded K-stride ASTR) ----
    // x part: vectorized b128 copies
    for (int i = tid; i < 128 * 8; i += 256) {
      int b = i >> 3, c = i & 7;
      *(bf16x8*)&As[b * ASTR + c * 8] =
          *(const bf16x8*)&Xb[(b * TT + t) * DD + c * 8];
    }
    if (t == 0) {
      bf16x8 zv = {};
      for (int i = tid; i < 128 * 32; i += 256)
        *(bf16x8*)&As[(i >> 5) * ASTR + DD + (i & 31) * 8] = zv;
    } else {
#if HAVE_TDM
      if (wave == 0) {
        // TDM: 2D tile 256 elems x 128 rows from H1seq[:, t-1, :] into the h-slots,
        // pad_interval=128 dwords (one 512B row), pad_amount=36 dwords (next row's
        // x slot + stride padding) -> lands exactly on the ASTR-strided layout.
        unsigned long long ga =
            (unsigned long long)(const void*)H1seq + (unsigned long long)(t - 1) * H1 * 2u;
        u32x4 g0 = {};
        g0[0] = 1u;                                        // count=1 (valid descriptor)
        g0[1] = AS_LDS_OFF + (unsigned)(DD * 2);           // lds_addr (h slot of row 0)
        g0[2] = (unsigned)(ga & 0xffffffffu);              // global_addr[31:0]
        g0[3] = (unsigned)((ga >> 32) & 0x01ffffffu) | (2u << 30);  // addr[56:32] | type=2
        i32x8 g1 = {};
        g1[0] = (1 << 16) | (1 << 20) | (6 << 22) | (35 << 25); // data_size=2B, pad en, 128dw/36dw
        g1[1] = (int)(256u << 16);                         // tensor_dim0 = 256
        g1[2] = (int)(128u << 16);                         // tensor_dim1 = 128
        g1[3] = (int)(256u << 16);                         // tile_dim0 = 256
        g1[4] = 128;                                       // tile_dim1 = 128
        g1[5] = (int)(TT * H1);                            // tensor_dim0_stride = T*H1
        i32x4 z4 = {};
#if defined(__clang_major__) && (__clang_major__ >= 23)
        i32x8 z8 = {};
        __builtin_amdgcn_tensor_load_to_lds(g0, g1, z4, z4, z8, 0);
#else
        __builtin_amdgcn_tensor_load_to_lds(g0, g1, z4, z4, 0);
#endif
#if __has_builtin(__builtin_amdgcn_s_wait_tensorcnt)
        __builtin_amdgcn_s_wait_tensorcnt(0);
#else
        asm volatile("s_wait_tensorcnt 0x0" ::: "memory");
#endif
      }
#else
      for (int i = tid; i < 128 * 32; i += 256) {
        int b = i >> 5, c = i & 31;
        *(bf16x8*)&As[b * ASTR + DD + c * 8] =
            *(const bf16x8*)&H1seq[(b * TT + (t - 1)) * H1 + c * 8];
      }
#endif
    }
    __syncthreads();

    f32x8 acc[8] = {};   // 8 coltiles of 16 (i,i,f,f,g,g,o,o)
    #pragma unroll 1
    for (int ks = 0; ks < 10; ++ks) {
      // A fragment: two contiguous 16B chunks (ds_load_b128 x2)
      const int ak = ks * 32 + half * 8;
      bf16x16 afrag = cat8(*(const bf16x8*)&As[arow * ASTR + ak],
                           *(const bf16x8*)&As[arow * ASTR + ak + 16]);
      #pragma unroll
      for (int ct = 0; ct < 8; ++ct) {
        const int bcol = ct * 16 + ln;
        const int bk = ks * 32 + half * 8;
        bf16x16 bfrag = cat8(*(const bf16x8*)&Bs[bcol * BSTR + bk],
                             *(const bf16x8*)&Bs[bcol * BSTR + bk + 16]);
        acc[ct] = __builtin_amdgcn_wmma_f32_16x16x32_bf16(
            false, afrag, false, bfrag, (short)0, acc[ct], false, false);
      }
    }

    // Gate nonlinearity + state update; lane owns rows rbase+half*8+v, units ln, ln+16.
    #pragma unroll
    for (int v = 0; v < 8; ++v) {
      int m = rbase + half * 8 + v;
      #pragma unroll
      for (int pr = 0; pr < 2; ++pr) {
        int u = ln + pr * 16;
        float iv = sigf (acc[0 + pr][v] + b1s[ 0 + u]);
        float fv = sigf (acc[2 + pr][v] + b1s[32 + u]);
        float gv = tanhf(acc[4 + pr][v] + b1s[64 + u]);
        float ov = sigf (acc[6 + pr][v] + b1s[96 + u]);
        float c  = fv * Cs[m * 32 + u] + iv * gv;
        Cs[m * 32 + u] = c;
        float h  = ov * tanhf(c);
        H1seq[(m * TT + t) * H1 + w * 32 + u] = f2bf(h);
      }
    }

    // Grid-wide barrier: all WGs see H1seq[t] before step t+1.
    __syncthreads();
    if (tid == 0) {
      __threadfence();
      unsigned g = __hip_atomic_load(&sync[1], __ATOMIC_ACQUIRE, __HIP_MEMORY_SCOPE_AGENT);
      unsigned prev = __hip_atomic_fetch_add(&sync[0], 1u, __ATOMIC_ACQ_REL, __HIP_MEMORY_SCOPE_AGENT);
      if (prev == NWG1 - 1) {
        __hip_atomic_store(&sync[0], 0u, __ATOMIC_RELAXED, __HIP_MEMORY_SCOPE_AGENT);
        __hip_atomic_fetch_add(&sync[1], 1u, __ATOMIC_ACQ_REL, __HIP_MEMORY_SCOPE_AGENT);
      } else {
        while (__hip_atomic_load(&sync[1], __ATOMIC_ACQUIRE, __HIP_MEMORY_SCOPE_AGENT) == g)
          __builtin_amdgcn_s_sleep(4);
      }
    }
    __syncthreads();
  }
}

// ---------------- K2: XZ2 = H1seq @ W2 + b2 (WMMA bf16, M=131072,N=64,K=256) ----------------
__global__ void __launch_bounds__(256, 1)
k_xz2(const unsigned short* __restrict__ H1seq, const unsigned short* __restrict__ W2bt,
      const float* __restrict__ b2, float* __restrict__ XZ2) {
  const int tid = threadIdx.x, lane = tid & 31, wave = tid >> 5;
  const int half = lane >> 4, ln = lane & 15;
  const int rbase = blockIdx.x * 128 + wave * 16;
  const int arow = rbase + ln;

  f32x8 acc[4] = {};
  #pragma unroll 1
  for (int ks = 0; ks < 8; ++ks) {
    __builtin_prefetch(&H1seq[arow * H1 + ((ks + 1) & 7) * 32], 0, 1);
    const int ak = ks * 32 + half * 8;
    bf16x16 afrag = cat8(*(const bf16x8*)&H1seq[arow * H1 + ak],
                         *(const bf16x8*)&H1seq[arow * H1 + ak + 16]);
    #pragma unroll
    for (int ct = 0; ct < 4; ++ct) {
      const int col = ct * 16 + ln;
      const int bk = ks * 32 + half * 8;
      bf16x16 bfrag = cat8(*(const bf16x8*)&W2bt[col * H1 + bk],
                           *(const bf16x8*)&W2bt[col * H1 + bk + 16]);
      acc[ct] = __builtin_amdgcn_wmma_f32_16x16x32_bf16(
          false, afrag, false, bfrag, (short)0, acc[ct], false, false);
    }
  }
  #pragma unroll
  for (int ct = 0; ct < 4; ++ct) {
    int col = ct * 16 + ln;
    float bias = b2[col];
    #pragma unroll
    for (int v = 0; v < 8; ++v) {
      int row = rbase + half * 8 + v;
      XZ2[row * (4 * H2) + col] = acc[ct][v] + bias;
    }
  }
}

// ---------------- K3: LSTM2 recurrence (hidden=16), single WG ----------------
__global__ void k_lstm2(const float* __restrict__ XZ2, const float* __restrict__ U2,
                        float* __restrict__ emb) {
  __shared__ float zb[128 * 64];
  __shared__ float h2s[128 * 16], c2s[128 * 16];
  __shared__ float U2s[16 * 64];
  const int tid = threadIdx.x;
  for (int i = tid; i < 16 * 64; i += 256) U2s[i] = U2[i];
  for (int i = tid; i < 128 * 16; i += 256) { h2s[i] = 0.f; c2s[i] = 0.f; }
  __syncthreads();
  for (int t = 0; t < TT; ++t) {
    for (int i = tid; i < 128 * 64; i += 256) {
      int b = i >> 6, n = i & 63;
      float z = XZ2[(b * TT + t) * 64 + n];
      #pragma unroll
      for (int k = 0; k < 16; ++k) z += h2s[b * 16 + k] * U2s[k * 64 + n];
      zb[i] = z;
    }
    __syncthreads();
    for (int i = tid; i < 128 * 16; i += 256) {
      int b = i >> 4, u = i & 15;
      float iv = sigf (zb[b * 64 +       u]);
      float fv = sigf (zb[b * 64 + 16 +  u]);
      float gv = tanhf(zb[b * 64 + 32 +  u]);
      float ov = sigf (zb[b * 64 + 48 +  u]);
      float c = fv * c2s[i] + iv * gv; c2s[i] = c;
      float h = ov * tanhf(c);         h2s[i] = h;
      emb[(b * TT + t) * H2 + u] = h;
    }
    __syncthreads();
  }
}

// ---------------- K4: Koopman operator MLPs + Jordan evolution + lin-loss partials ----------------
__global__ void k_jordan(const float* __restrict__ emb,
                         const float* __restrict__ Ar0, const float* __restrict__ br0,
                         const float* __restrict__ Ar1, const float* __restrict__ br1,
                         const float* __restrict__ Ac0, const float* __restrict__ bc0,
                         const float* __restrict__ Ac1, const float* __restrict__ bc1,
                         float* __restrict__ jordan, float* __restrict__ linpart) {
  __shared__ float sAr0[RR * KAUX], sAr1[KAUX * RR], sAc0[2 * PP * KAUX], sAc1[KAUX * 2 * PP];
  __shared__ float sbr0[KAUX], sbc0[KAUX], sbr1[RR], sbc1[2 * PP];
  __shared__ float red[256];
  const int tid = threadIdx.x;
  for (int i = tid; i < RR * KAUX; i += 256) { sAr0[i] = Ar0[i]; sAc0[i] = Ac0[i]; }
  for (int i = tid; i < KAUX * RR; i += 256) { sAr1[i] = Ar1[i]; sAc1[i] = Ac1[i]; }
  for (int i = tid; i < KAUX; i += 256)      { sbr0[i] = br0[i]; sbc0[i] = bc0[i]; }
  if (tid < RR)     sbr1[tid] = br1[tid];
  if (tid < 2 * PP) sbc1[tid] = bc1[tid];
  __syncthreads();

  const int row = blockIdx.x * 256 + tid;     // (b*T + t)
  float e[16];
  #pragma unroll
  for (int k = 0; k < 16; ++k) e[k] = emb[row * 16 + k];

  float ar[RR] = {}, ac[2 * PP] = {};
  for (int h = 0; h < KAUX; ++h) {
    float hr = sbr0[h], hc = sbc0[h];
    #pragma unroll
    for (int k = 0; k < RR; ++k) { hr += e[k] * sAr0[k * KAUX + h]; hc += e[RR + k] * sAc0[k * KAUX + h]; }
    hr = leaky(hr); hc = leaky(hc);
    #pragma unroll
    for (int o = 0; o < RR; ++o) { ar[o] += hr * sAr1[h * RR + o]; ac[o] += hc * sAc1[h * RR + o]; }
  }
  float jr[16];
  #pragma unroll
  for (int k = 0; k < RR; ++k) jr[k] = __expf(leaky(ar[k] + sbr1[k]) * DT_C) * e[k];
  #pragma unroll
  for (int p = 0; p < PP; ++p) {
    float mu = leaky(ac[2 * p] + sbc1[2 * p]);
    float th = leaky(ac[2 * p + 1] + sbc1[2 * p + 1]);
    float s = __expf(mu * DT_C);
    float e1 = s * __cosf(th * DT_C), e2 = s * __sinf(th * DT_C);
    jr[RR + 2 * p]     = (e1 + e2) * e[2 * p];
    jr[RR + 2 * p + 1] = (e1 - e2) * e[2 * p + 1];
  }
  #pragma unroll
  for (int k = 0; k < 16; ++k) jordan[row * 16 + k] = jr[k];

  // lin_loss partial: (emb[b, t+1] - jordan[b, t])^2, t < T-1
  float lin = 0.f;
  if ((row % TT) < TT - 1) {
    #pragma unroll
    for (int k = 0; k < 16; ++k) { float d = emb[(row + 1) * 16 + k] - jr[k]; lin += d * d; }
  }
  red[tid] = lin; __syncthreads();
  for (int s = 128; s > 0; s >>= 1) { if (tid < s) red[tid] += red[tid + s]; __syncthreads(); }
  if (tid == 0) linpart[blockIdx.x] = red[0];
}

// ---------------- K5: decoder (applied to jordan & emb) + recon partials ----------------
__global__ void __launch_bounds__(256, 1)
k_dec(const float* __restrict__ X, const float* __restrict__ jordan, const float* __restrict__ emb,
      const float* __restrict__ Wd0, const float* __restrict__ bd0,
      const float* __restrict__ Wd1, const float* __restrict__ bd1,
      float* __restrict__ out, float* __restrict__ rsspart, float* __restrict__ xsspart) {
  extern __shared__ float sm[];
  float* sW0 = sm;                       // 16*256
  float* sW1 = sW0 + H2 * DE;            // 256*64
  float* sb0 = sW1 + DE * DD;            // 256
  float* sb1 = sb0 + DE;                 // 64
  float* red = sb1 + DD;                 // 256
  const int tid = threadIdx.x;
  for (int i = tid; i < H2 * DE; i += 256) sW0[i] = Wd0[i];
  for (int i = tid; i < DE * DD; i += 256) sW1[i] = Wd1[i];
  for (int i = tid; i < DE; i += 256)      sb0[i] = bd0[i];
  if (tid < DD) sb1[tid] = bd1[tid];
  __syncthreads();

  const long row = (long)blockIdx.x * 256 + tid;
  float x[DD];
  #pragma unroll
  for (int k = 0; k < DD; ++k) x[k] = X[row * DD + k];

  float o[DD], z[H2];
  float rss = 0.f, xss = 0.f;
  for (int pass = 0; pass < 2; ++pass) {
    const float* src = pass == 0 ? jordan : emb;
    #pragma unroll
    for (int k = 0; k < H2; ++k) z[k] = src[row * H2 + k];
    #pragma unroll
    for (int k = 0; k < DD; ++k) o[k] = 0.f;
    for (int h = 0; h < DE; ++h) {
      float hv = sb0[h];
      #pragma unroll
      for (int k = 0; k < H2; ++k) hv += z[k] * sW0[k * DE + h];
      hv = leaky(hv);
      #pragma unroll
      for (int k = 0; k < DD; ++k) o[k] += hv * sW1[h * DD + k];
    }
    if (pass == 0) {
      #pragma unroll
      for (int k = 0; k < DD; ++k) out[row * DD + k] = leaky(o[k] + sb1[k]) + x[k];
    } else {
      #pragma unroll
      for (int k = 0; k < DD; ++k) {
        float rc = leaky(o[k] + sb1[k]);
        float d = rc - x[k];
        rss += d * d; xss += x[k] * x[k];
      }
    }
  }
  red[tid] = rss; __syncthreads();
  for (int s = 128; s > 0; s >>= 1) { if (tid < s) red[tid] += red[tid + s]; __syncthreads(); }
  if (tid == 0) rsspart[blockIdx.x] = red[0];
  __syncthreads();
  red[tid] = xss; __syncthreads();
  for (int s = 128; s > 0; s >>= 1) { if (tid < s) red[tid] += red[tid + s]; __syncthreads(); }
  if (tid == 0) xsspart[blockIdx.x] = red[0];
}

// ---------------- K6: final loss reduction ----------------
__global__ void k_loss(const float* __restrict__ linpart, const float* __restrict__ rsspart,
                       const float* __restrict__ xsspart, float* __restrict__ out) {
  __shared__ float r0[256], r1[256], r2[256];
  const int tid = threadIdx.x;
  float a = 0.f, b = 0.f, c = 0.f;
  for (int i = tid; i < 512; i += 256) { a += linpart[i]; b += rsspart[i]; c += xsspart[i]; }
  r0[tid] = a; r1[tid] = b; r2[tid] = c; __syncthreads();
  for (int s = 128; s > 0; s >>= 1) {
    if (tid < s) { r0[tid] += r0[tid + s]; r1[tid] += r1[tid + s]; r2[tid] += r2[tid + s]; }
    __syncthreads();
  }
  if (tid == 0) {
    float denom = (float)(TT - 1) * (float)BB;
    float mean_in = r2[0] / (float)((long)BB * TT * DD);
    float recon = (r1[0] / mean_in) / denom;
    float lin = r0[0] / denom;
    out[(long)BT * DD] = RECON_HP_C * recon + lin;
  }
}

// ---------------- host: launch sequence ----------------
extern "C" void kernel_launch(void* const* d_in, const int* in_sizes, int n_in,
                              void* d_out, int out_size, void* d_ws, size_t ws_size,
                              hipStream_t stream) {
  const float* X   = (const float*)d_in[0];
  const float* W1  = (const float*)d_in[1];
  const float* U1  = (const float*)d_in[2];
  const float* b1  = (const float*)d_in[3];
  const float* W2  = (const float*)d_in[4];
  const float* U2  = (const float*)d_in[5];
  const float* b2  = (const float*)d_in[6];
  const float* Ar0 = (const float*)d_in[7];
  const float* br0 = (const float*)d_in[8];
  const float* Ar1 = (const float*)d_in[9];
  const float* br1 = (const float*)d_in[10];
  const float* Ac0 = (const float*)d_in[11];
  const float* bc0 = (const float*)d_in[12];
  const float* Ac1 = (const float*)d_in[13];
  const float* bc1 = (const float*)d_in[14];
  const float* Wd0 = (const float*)d_in[15];
  const float* bd0 = (const float*)d_in[16];
  const float* Wd1 = (const float*)d_in[17];
  const float* bd1 = (const float*)d_in[18];

  // carve workspace (256B aligned)
  char* p = (char*)d_ws;
  auto take = [&](size_t bytes) { char* r = p; p += (bytes + 255) & ~size_t(255); return r; };
  unsigned short* Xb   = (unsigned short*)take((size_t)BT * DD * 2);
  unsigned short* W1b  = (unsigned short*)take((size_t)DD * 4 * H1 * 2);
  unsigned short* U1b  = (unsigned short*)take((size_t)H1 * 4 * H1 * 2);
  unsigned short* W2bt = (unsigned short*)take((size_t)H1 * 4 * H2 * 2);
  unsigned short* H1s  = (unsigned short*)take((size_t)BT * H1 * 2);
  float*          XZ2  = (float*)take((size_t)BT * 4 * H2 * 4);
  float*          emb  = (float*)take((size_t)BT * H2 * 4);
  float*          jor  = (float*)take((size_t)BT * H2 * 4);
  float*          lin  = (float*)take(512 * 4);
  float*          rssp = (float*)take(512 * 4);
  float*          xssp = (float*)take(512 * 4);
  unsigned*       sync = (unsigned*)take(64);

  float* out = (float*)d_out;

  k_prep<<<2048, 256, 0, stream>>>(X, W1, U1, W2, Xb, W1b, U1b, W2bt, sync);

  size_t lds1 = (size_t)(128 * BSTR * 2 + 128 * ASTR * 2) + (size_t)(128 * 32 + 128) * 4;
  k_lstm1<<<NWG1, 256, lds1, stream>>>(b1, Xb, W1b, U1b, H1s, sync);

  k_xz2<<<BT / 128, 256, 0, stream>>>(H1s, W2bt, b2, XZ2);

  k_lstm2<<<1, 256, 0, stream>>>(XZ2, U2, emb);

  k_jordan<<<BT / 256, 256, 0, stream>>>(emb, Ar0, br0, Ar1, br1, Ac0, bc0, Ac1, bc1, jor, lin);

  size_t lds5 = (size_t)(H2 * DE + DE * DD + DE + DD + 256) * 4;
  k_dec<<<BT / 256, 256, lds5, stream>>>(X, jor, emb, Wd0, bd0, Wd1, bd1, out, rssp, xssp);

  k_loss<<<1, 256, 0, stream>>>(lin, rssp, xssp, out);
}
